// KNRM_47691316855433
// MI455X (gfx1250) — matrix operand order repaired and checked by reading, not compiled, not run on hardware
//
#include <hip/hip_runtime.h>
#include <math.h>

typedef _Float16 v16h __attribute__((ext_vector_type(16)));
typedef _Float16 h8   __attribute__((ext_vector_type(8)));
typedef float    v8f  __attribute__((ext_vector_type(8)));

#define LQ   30
#define LQP  32
#define LDOC 128
#define DIM  50
#define KPAD 64
#define KN   21
#define MMS  129   // mmbuf row stride (odd -> no LDS bank conflicts across rows)

__global__ __launch_bounds__(256) void knrm_fused(
    const int* __restrict__ q1, const int* __restrict__ d1,
    const int* __restrict__ q2, const int* __restrict__ d2,
    const float* __restrict__ emb,
    const float* __restrict__ mlp_w, const float* __restrict__ mlp_b,
    float* __restrict__ out)
{
  __shared__ __align__(16) _Float16 lq[LQP * KPAD];    // 4 KB  (query rows, f16, K-padded)
  __shared__ __align__(16) _Float16 ldm[LDOC * KPAD];  // 16 KB (doc rows)
  __shared__ float mmbuf[LQ * MMS];                    // 15.5 KB cosine match matrix
  __shared__ int   qtok[LQ];
  __shared__ int   dtok[LDOC];
  __shared__ float kacc[KN];
  __shared__ float lgts[2];

  const int tid  = threadIdx.x;
  const int lane = tid & 31;
  const int wave = tid >> 5;
  const int b    = blockIdx.x;

  for (int p = 0; p < 2; ++p) {
    const int* qp = (p == 0) ? q1 : q2;
    const int* dp = (p == 0) ? d1 : d2;

    __syncthreads();                       // protect LDS reuse across pairs
    if (tid < KN)   kacc[tid] = 0.0f;
    if (tid < LQ)   qtok[tid] = qp[b * LQ   + tid];
    if (tid < LDOC) dtok[tid] = dp[b * LDOC + tid];

    // ---------- phase 1: gather + L2-normalize -> f16 LDS tiles ----------
    for (int r = wave; r < LQP + LDOC; r += 8) {
      _Float16* rowp = (r < LQP) ? &lq[r * KPAD] : &ldm[(r - LQP) * KPAD];
      int tok = (r < LQ) ? qp[b * LQ + r]
                         : ((r < LQP) ? -1 : dp[b * LDOC + (r - LQP)]);
      float e0 = 0.0f, e1 = 0.0f;
      if (tok >= 0) {
        const float* erow = emb + (size_t)tok * DIM;
        e0 = erow[lane];
        if (lane + 32 < DIM) e1 = erow[lane + 32];
      }
      float s = e0 * e0 + e1 * e1;
      for (int msk = 16; msk > 0; msk >>= 1) s += __shfl_xor(s, msk, 32);
      float inv = 1.0f / fmaxf(sqrtf(s), 1e-12f);
      if (tok < 0) inv = 0.0f;               // zero-fill padding rows 30,31
      rowp[lane]      = (_Float16)(e0 * inv);
      rowp[lane + 32] = (lane + 32 < DIM) ? (_Float16)(e1 * inv) : (_Float16)0.0f;
    }
    __syncthreads();

    // ---------- phase 2: cosine match matrix via v_wmma_f32_16x16x32_f16 ----------
    {
      const int m  = lane & 15;
      const int hi = lane >> 4;
      const int nt = wave;                                   // N-tile 0..7
      const _Float16* brow = &ldm[(nt * 16 + m) * KPAD];     // B[k][n] = ed[n][k]
      #pragma unroll
      for (int mt = 0; mt < 2; ++mt) {
        const _Float16* arow = &lq[(mt * 16 + m) * KPAD];
        v8f c = {};
        #pragma unroll
        for (int ks = 0; ks < KPAD; ks += 32) {
          // A 16x32 f16, contiguous k order used here (same as B): elem h -> K = ks + hi*16 + h at row m = lane&15
          h8 a0 = *(const h8*)(arow + ks + hi * 16);
          h8 a1 = *(const h8*)(arow + ks + hi * 16 + 8);
          // B 32x16 f16, same contiguous k order as A: elem h -> K = ks + hi*16 + h at column n = lane&15
          h8 b0 = *(const h8*)(brow + ks + hi * 16);
          h8 b1 = *(const h8*)(brow + ks + hi * 16 + 8);
          v16h av = __builtin_shufflevector(a0, a1, 0,1,2,3,4,5,6,7,8,9,10,11,12,13,14,15);
          v16h bv = __builtin_shufflevector(b0, b1, 0,1,2,3,4,5,6,7,8,9,10,11,12,13,14,15);
          c = __builtin_amdgcn_wmma_f32_16x16x32_f16(false, av, false, bv,
                                                     (short)0, c, false, false);
        }
        #pragma unroll
        for (int r = 0; r < 8; ++r) {
          int mg = mt * 16 + hi * 8 + r;          // C layout: VGPR r -> M = r + hi*8
          if (mg < LQ) mmbuf[mg * MMS + nt * 16 + m] = c[r];
        }
      }
    }
    __syncthreads();

    // ---------- phase 3: 21 RBF kernels, soft-TF pooling ----------
    for (int t = tid; t < KN * LQ; t += 256) {
      int i  = t / LQ;
      int m2 = t - i * LQ;
      float S = 0.0f;
      if (i < 20) {
        const float mu = 0.1f * (float)i - 0.95f;
        const float C2 = -72.13475204444817f;     // -1/(2*0.1^2) * log2(e)
        #pragma unroll 4
        for (int n = 0; n < LDOC; ++n) {
          float x = mmbuf[m2 * MMS + n] - mu;
          S += __builtin_amdgcn_exp2f(C2 * x * x);
        }
      } else {
        // exact-match kernel (sigma=1e-3): delta at cosine==1 == identical tokens
        int qt = qtok[m2], cnt = 0;
        for (int n = 0; n < LDOC; ++n) cnt += (dtok[n] == qt) ? 1 : 0;
        S = (float)cnt;
      }
      atomicAdd(&kacc[i], __logf(1.0f + S));
    }
    __syncthreads();

    if (tid == 0) {
      float acc = mlp_b[0];
      #pragma unroll
      for (int i = 0; i < KN; ++i) acc += kacc[i] * mlp_w[i];
      lgts[p] = acc;
    }
  }
  __syncthreads();
  if (tid == 0) {
    float z = lgts[0] - lgts[1];
    out[b] = 1.0f / (1.0f + __expf(-z));
  }
}

extern "C" void kernel_launch(void* const* d_in, const int* in_sizes, int n_in,
                              void* d_out, int out_size, void* d_ws, size_t ws_size,
                              hipStream_t stream) {
  const int*   q1  = (const int*)d_in[0];
  const int*   d1  = (const int*)d_in[1];
  const int*   q2  = (const int*)d_in[2];
  const int*   d2  = (const int*)d_in[3];
  const float* emb = (const float*)d_in[4];
  const float* w   = (const float*)d_in[5];
  const float* bia = (const float*)d_in[6];
  float* out = (float*)d_out;

  int B = in_sizes[0] / LQ;   // 4096
  knrm_fused<<<B, 256, 0, stream>>>(q1, d1, q2, d2, emb, w, bia, out);
}
